// Teacher_87625922773503
// MI455X (gfx1250) — matrix operand order repaired and checked by reading, not compiled, not run on hardware
//
#include <hip/hip_runtime.h>
#include <hip/hip_bf16.h>
#include <math.h>

// ---------------------------------------------------------------------------
// CDNA5 (gfx1250) seq2seq LSTM:
//   1) gather+convert inputs to f16
//   2) GEMM1 (WMMA f16, 32x32 register-blocked): Gpre[t][4H] = Wih@x_t + b
//   3) persistent recurrence kernel (16 WGs, grid barrier, double-buffered h)
//   4) GEMM2 (WMMA f16, 32x32 register-blocked): logits[t][Vout] = Wout@h_t + bout
//   5) row softmax
// ---------------------------------------------------------------------------

typedef __attribute__((ext_vector_type(16))) _Float16 v16h;
typedef __attribute__((ext_vector_type(8)))  float    v8f;

#define NWG 16           // workgroups in the recurrence kernel (must all be resident)

// ------------------------------- utilities ---------------------------------

__global__ void k_init(_Float16* hbuf, unsigned* syncbuf, int H) {
    int i = blockIdx.x * blockDim.x + threadIdx.x;
    for (int k = i; k < 2 * H; k += gridDim.x * blockDim.x) hbuf[k] = (_Float16)0.f;
    if (i < 64) syncbuf[i] = 0u;
}

__global__ void k_f32_to_f16(const float* __restrict__ src, _Float16* __restrict__ dst, long n) {
    long i = (long)blockIdx.x * blockDim.x + threadIdx.x;
    if (i < n) dst[i] = (_Float16)src[i];
}

// X16[t][e] = (f16) emb[state[t]][e]
__global__ void k_gather_emb(const int* __restrict__ state, const float* __restrict__ emb,
                             _Float16* __restrict__ X16, int E, int Vin) {
    int t = blockIdx.x;
    int tok = state[t];
    if (tok < 0) tok = 0;
    if (tok >= Vin) tok = Vin - 1;
    const float* src = emb + (size_t)tok * E;
    _Float16* dst = X16 + (size_t)t * E;
    for (int e = threadIdx.x; e < E; e += blockDim.x) dst[e] = (_Float16)src[e];
}

// ------------------------------ WMMA GEMM ----------------------------------
// C[n*M + m] = sum_k A[m][k]*B[n][k] + bias0[m] (+ bias1[m])
// A: [M][K] f16 row-major, B: [N][K] f16 row-major (logical B = K x N with
// column n = row n of Bsrc). One wave per 32x32 output block (2x2 WMMA tiles)
// so each loaded A/B fragment is reused twice: 4 WMMAs per 4 fragment loads.

// Load a 16x32 A fragment: contiguous K order, lane half hi holds K = hi*16 + 0..15 (VGPR v holds K pairs)
__device__ inline v16h load_afrag(const _Float16* __restrict__ arow, int kb, int hi) {
    v16h a;
    const _Float16* ap = arow + kb;
    #pragma unroll
    for (int v = 0; v < 8; ++v) {
        int k0 = (hi << 4) + (v << 1);  // hi*16+v*2
        a[2 * v]     = ap[k0];
        a[2 * v + 1] = ap[k0 + 1];
    }
    return a;
}

// Load a 32x16 B fragment: lane holds column n, 16 sequential K halves
__device__ inline v16h load_bfrag(const _Float16* __restrict__ brow, int kb, int hi) {
    v16h b;
    const _Float16* bp = brow + kb + (hi << 4);
    #pragma unroll
    for (int e = 0; e < 16; ++e) b[e] = bp[e];
    return b;
}

__global__ void k_wmma_gemm(const _Float16* __restrict__ A, const _Float16* __restrict__ B,
                            float* __restrict__ C,
                            const float* __restrict__ bias0, const float* __restrict__ bias1,
                            int M, int N, int K) {
    const int lane = threadIdx.x & 31;          // wave32
    const int lo   = lane & 15;
    const int hi   = lane >> 4;

    const int Mblk = (M + 31) >> 5;             // 32-wide M blocks
    const int Nblk = (N + 31) >> 5;             // 32-wide N blocks
    long wid = (long)blockIdx.x * (blockDim.x >> 5) + (threadIdx.x >> 5);
    if (wid >= (long)Mblk * Nblk) return;
    const int mb = (int)(wid % Mblk);
    const int nb = (int)(wid / Mblk);
    const int mbase = mb << 5;
    const int nbase = nb << 5;

    // clamped per-lane source rows (stores are guarded; loads only need bounds)
    int am0 = mbase + lo;       if (am0 >= M) am0 = M - 1;
    int am1 = mbase + 16 + lo;  if (am1 >= M) am1 = M - 1;
    int bn0 = nbase + lo;       if (bn0 >= N) bn0 = N - 1;
    int bn1 = nbase + 16 + lo;  if (bn1 >= N) bn1 = N - 1;
    const _Float16* arow0 = A + (size_t)am0 * K;
    const _Float16* arow1 = A + (size_t)am1 * K;
    const _Float16* brow0 = B + (size_t)bn0 * K;
    const _Float16* brow1 = B + (size_t)bn1 * K;

    v8f c00 = {0.f,0.f,0.f,0.f,0.f,0.f,0.f,0.f};
    v8f c01 = c00, c10 = c00, c11 = c00;

    for (int kb = 0; kb < K; kb += 32) {
        v16h a0 = load_afrag(arow0, kb, hi);
        v16h a1 = load_afrag(arow1, kb, hi);
        v16h b0 = load_bfrag(brow0, kb, hi);
        v16h b1 = load_bfrag(brow1, kb, hi);
        c00 = __builtin_amdgcn_wmma_f32_16x16x32_f16(false, a0, false, b0, (short)0, c00, false, false);
        c01 = __builtin_amdgcn_wmma_f32_16x16x32_f16(false, a0, false, b1, (short)0, c01, false, false);
        c10 = __builtin_amdgcn_wmma_f32_16x16x32_f16(false, a1, false, b0, (short)0, c10, false, false);
        c11 = __builtin_amdgcn_wmma_f32_16x16x32_f16(false, a1, false, b1, (short)0, c11, false, false);
    }

    // ---- store: C/D layout — VGPR g: lanes0-15 M=g, lanes16-31 M=g+8 ----
    #pragma unroll
    for (int half = 0; half < 2; ++half) {            // N sub-tile
        const int col = nbase + (half << 4) + lo;
        if (col >= N) continue;
        float* ccol = C + (size_t)col * M;
        #pragma unroll
        for (int mh = 0; mh < 2; ++mh) {              // M sub-tile
            const v8f& acc = (mh == 0) ? (half == 0 ? c00 : c01)
                                       : (half == 0 ? c10 : c11);
            #pragma unroll
            for (int g = 0; g < 8; ++g) {
                int row = mbase + (mh << 4) + g + (hi << 3);
                if (row < M) {
                    float v = acc[g];
                    if (bias0) v += bias0[row];
                    if (bias1) v += bias1[row];
                    ccol[row] = v;
                }
            }
        }
    }
}

// --------------------------- persistent recurrence -------------------------

__device__ inline float sigmoidf_(float x) { return 1.f / (1.f + expf(-x)); }

__device__ inline void grid_sync(unsigned* sb, unsigned nwg) {
    __syncthreads();
    __threadfence();                       // release: h stores visible agent-wide
    if (threadIdx.x == 0) {
        volatile unsigned* vs = sb;
        unsigned g = vs[1];                // read generation BEFORE arriving
        if (atomicAdd(&sb[0], 1u) == nwg - 1u) {
            vs[0] = 0u;
            __threadfence();
            atomicAdd(&sb[1], 1u);         // bump generation, wake spinners
        } else {
            while (vs[1] == g) { __builtin_amdgcn_s_sleep(1); }
        }
    }
    __syncthreads();
    __threadfence();                       // acquire: invalidate stale h lines
}

// 16 WGs x 256 threads. WG g owns hidden units j in [g*64, g*64+64).
// Thread tid: gate q = tid/64, unit jj = tid%64 -> Whh row r = q*H + j.
__global__ void k_lstm_recurrence(const _Float16* __restrict__ WhhE,
                                  const _Float16* __restrict__ WhhD,
                                  const float* __restrict__ Gpre,     // [S][4H]
                                  const float* __restrict__ dbih,
                                  const float* __restrict__ dbhh,
                                  _Float16* __restrict__ hbuf,        // [2][H] ping-pong
                                  _Float16* __restrict__ Hdec,        // [T][H]
                                  unsigned* __restrict__ syncbuf,
                                  int S, int T, int H) {
    __shared__ float hs[1024];     // h_{t-1} in f32
    __shared__ float gbuf[256];    // raw gate pre-activations for this WG

    const int tid = threadIdx.x;
    const int wg  = blockIdx.x;
    const int JPW = H / NWG;       // 64 units per WG
    const int q   = tid / JPW;     // gate index 0..3 (i,f,g,o)
    const int jj  = tid % JPW;
    const int j   = wg * JPW + jj; // hidden unit index
    const int r   = q * H + j;     // row in 4H gate vector

    const _Float16* wrowE = WhhE + (size_t)r * H;
    const _Float16* wrowD = WhhD + (size_t)r * H;
    const float dec_bias  = dbih[r] + dbhh[r];

    float creg = 0.f;              // c[j], owned by thread tid<JPW
    int parity = 0;
    const int total = S + T;

    for (int t = 0; t < total; ++t) {
        // stage h_{t-1} into LDS (f32)
        const _Float16* hin = hbuf + (size_t)parity * H;
        for (int k = tid; k < H; k += blockDim.x) hs[k] = (float)hin[k];
        __syncthreads();

        const bool enc = (t < S);
        float acc = enc ? Gpre[(size_t)t * 4 * H + r] : dec_bias;
        const _Float16* wrow = enc ? wrowE : wrowD;
        // keep next step's Gpre row warm in cache while we do the matvec
        if (enc && t + 1 < S) __builtin_prefetch(&Gpre[(size_t)(t + 1) * 4 * H + r], 0, 1);
        #pragma unroll 8
        for (int k = 0; k < H; ++k) acc += (float)wrow[k] * hs[k];
        gbuf[tid] = acc;
        __syncthreads();

        if (tid < JPW) {
            float iv = sigmoidf_(gbuf[tid]);
            float fv = sigmoidf_(gbuf[JPW + tid]);
            float gv = tanhf   (gbuf[2 * JPW + tid]);
            float ov = sigmoidf_(gbuf[3 * JPW + tid]);
            creg = fv * creg + iv * gv;
            float hv = ov * tanhf(creg);
            hbuf[(size_t)(parity ^ 1) * H + j] = (_Float16)hv;
            if (!enc) Hdec[(size_t)(t - S) * H + j] = (_Float16)hv;
        }
        grid_sync(syncbuf, NWG);
        parity ^= 1;
    }
}

// -------------------------------- softmax ----------------------------------

__global__ void k_softmax_rows(float* __restrict__ out, int V) {
    __shared__ float red[256];
    float* x = out + (size_t)blockIdx.x * V;
    const int tid = threadIdx.x;

    float m = -INFINITY;
    for (int i = tid; i < V; i += blockDim.x) m = fmaxf(m, x[i]);
    red[tid] = m; __syncthreads();
    for (int s = 128; s > 0; s >>= 1) {
        if (tid < s) red[tid] = fmaxf(red[tid], red[tid + s]);
        __syncthreads();
    }
    m = red[0]; __syncthreads();

    float sum = 0.f;
    for (int i = tid; i < V; i += blockDim.x) {
        float e = expf(x[i] - m);
        x[i] = e;
        sum += e;
    }
    red[tid] = sum; __syncthreads();
    for (int s = 128; s > 0; s >>= 1) {
        if (tid < s) red[tid] += red[tid + s];
        __syncthreads();
    }
    float inv = 1.f / red[0];
    for (int i = tid; i < V; i += blockDim.x) x[i] *= inv;
}

// -------------------------------- launcher ---------------------------------

extern "C" void kernel_launch(void* const* d_in, const int* in_sizes, int n_in,
                              void* d_out, int out_size, void* d_ws, size_t ws_size,
                              hipStream_t stream) {
    // Input order: state, max_len, emb, enc_Wih, enc_Whh, enc_bih, enc_bhh,
    //              dec_Wih, dec_Whh, dec_bih, dec_bhh, Wout, bout
    const int*   state   = (const int*)d_in[0];
    const float* emb     = (const float*)d_in[2];
    const float* enc_Wih = (const float*)d_in[3];
    const float* enc_Whh = (const float*)d_in[4];
    const float* enc_bih = (const float*)d_in[5];
    const float* enc_bhh = (const float*)d_in[6];
    const float* dec_Whh = (const float*)d_in[8];
    const float* dec_bih = (const float*)d_in[9];
    const float* dec_bhh = (const float*)d_in[10];
    const float* Wout    = (const float*)d_in[11];
    const float* bout    = (const float*)d_in[12];
    float* out = (float*)d_out;

    const int S    = in_sizes[0];               // 2048
    const int H    = in_sizes[5] / 4;           // 1024
    const int E    = in_sizes[3] / (4 * H);     // 512
    const int Vin  = in_sizes[2] / E;           // 32000
    const int Vout = in_sizes[12];              // 50257
    const int T    = out_size / Vout;           // 512 (max_len)

    // ---- carve workspace (256B aligned) ----
    char* p = (char*)d_ws;
    auto carve = [&](size_t bytes) -> void* {
        void* r = (void*)p;
        p += (bytes + 255) & ~(size_t)255;
        return r;
    };
    _Float16* X16     = (_Float16*)carve((size_t)S * E * 2);
    _Float16* Wih16   = (_Float16*)carve((size_t)4 * H * E * 2);
    _Float16* WhhE16  = (_Float16*)carve((size_t)4 * H * H * 2);
    _Float16* WhhD16  = (_Float16*)carve((size_t)4 * H * H * 2);
    _Float16* Wout16  = (_Float16*)carve((size_t)Vout * H * 2);
    float*    Gpre    = (float*)   carve((size_t)S * 4 * H * 4);
    _Float16* Hdec16  = (_Float16*)carve((size_t)T * H * 2);
    _Float16* hbuf    = (_Float16*)carve((size_t)2 * H * 2);
    unsigned* syncbuf = (unsigned*)carve(256);
    (void)ws_size; (void)n_in;

    // 1) init barrier state + h0 = 0
    k_init<<<1, 256, 0, stream>>>(hbuf, syncbuf, H);

    // 2) f32 -> f16 weight conversions
    {
        long n1 = (long)4 * H * E;
        k_f32_to_f16<<<(unsigned)((n1 + 255) / 256), 256, 0, stream>>>(enc_Wih, Wih16, n1);
        long n2 = (long)4 * H * H;
        k_f32_to_f16<<<(unsigned)((n2 + 255) / 256), 256, 0, stream>>>(enc_Whh, WhhE16, n2);
        k_f32_to_f16<<<(unsigned)((n2 + 255) / 256), 256, 0, stream>>>(dec_Whh, WhhD16, n2);
        long n3 = (long)Vout * H;
        k_f32_to_f16<<<(unsigned)((n3 + 255) / 256), 256, 0, stream>>>(Wout, Wout16, n3);
    }

    // 3) gather embeddings for the whole sequence
    k_gather_emb<<<S, 256, 0, stream>>>(state, emb, X16, E, Vin);

    // 4) GEMM1: Gpre[t][4H] = Wih @ x_t + (bih + bhh)   (M=4H, N=S, K=E)
    {
        int M = 4 * H, N = S, K = E;
        long blkcnt = (long)((M + 31) / 32) * ((N + 31) / 32);
        unsigned blocks = (unsigned)((blkcnt + 7) / 8);       // 8 waves / block
        k_wmma_gemm<<<blocks, 256, 0, stream>>>(Wih16, X16, Gpre, enc_bih, enc_bhh, M, N, K);
    }

    // 5) persistent recurrence: encoder (S steps) then decoder (T steps)
    k_lstm_recurrence<<<NWG, 256, 0, stream>>>(WhhE16, WhhD16, Gpre, dec_bih, dec_bhh,
                                               hbuf, Hdec16, syncbuf, S, T, H);

    // 6) GEMM2: logits[t][Vout] = Wout @ h_t + bout   (M=Vout, N=T, K=H)
    {
        int M = Vout, N = T, K = H;
        long blkcnt = (long)((M + 31) / 32) * ((N + 31) / 32);
        unsigned blocks = (unsigned)((blkcnt + 7) / 8);
        k_wmma_gemm<<<blocks, 256, 0, stream>>>(Wout16, Hdec16, out, bout, nullptr, M, N, K);
    }

    // 7) softmax over each of the T rows (in place on d_out)
    k_softmax_rows<<<T, 256, 0, stream>>>(out, Vout);
}
